// Encoder_Adversarial_GCN_55714315764099
// MI455X (gfx1250) — matrix operand order repaired and checked by reading, not compiled, not run on hardware
//
#include <hip/hip_runtime.h>

#define N_NODES 10000
#define IN_DIM  256
#define HID_DIM 512
#define OUT_DIM 256

typedef __bf16 bf16;
typedef __attribute__((ext_vector_type(16))) __bf16 v16bf;
typedef __attribute__((ext_vector_type(8)))  __bf16 v8bf;
typedef __attribute__((ext_vector_type(8)))  float  v8f;

// ---------------------------------------------------------------- utilities

__global__ void fill_f32(float* __restrict__ p, float v, int n) {
    int g = blockIdx.x * blockDim.x + threadIdx.x;
    if (g < n) p[g] = v;
}

__global__ void deg_count(const int* __restrict__ dst, float* __restrict__ deg, int E) {
    int g = blockIdx.x * blockDim.x + threadIdx.x;
    if (g < E) atomicAdd(&deg[dst[g]], 1.0f);
}

__global__ void rsqrt_inplace(float* __restrict__ p, int n) {
    int g = blockIdx.x * blockDim.x + threadIdx.x;
    if (g < n) p[g] = rsqrtf(p[g]);
}

__global__ void cvt_bf16(const float* __restrict__ in, bf16* __restrict__ out, int n) {
    int g = blockIdx.x * blockDim.x + threadIdx.x;
    if (g < n) out[g] = (bf16)in[g];
}

// W is [K][N] row-major; produce Wt [N][K] in bf16 so B-fragments are contiguous.
__global__ void transpose_cvt(const float* __restrict__ W, bf16* __restrict__ Wt,
                              int K, int N) {
    int g = blockIdx.x * blockDim.x + threadIdx.x;
    if (g < K * N) {
        int k = g / N;
        int n = g - k * N;
        Wt[(size_t)n * K + k] = (bf16)W[g];
    }
}

// ---------------------------------------------------------------- WMMA GEMM
// C[M,NDIM] = A[M,KDIM] * B[KDIM,NDIM], A bf16 row-major, B given transposed
// (Bt[NDIM][KDIM] bf16), C fp32. One wave computes a 16x64 strip: the A
// fragment is reused across 4 v_wmma_f32_16x16x32_bf16 issues per K-step.
template <int KDIM, int NDIM>
__global__ __launch_bounds__(128) void gemm_bf16_wmma(const bf16* __restrict__ A,
                                                      const bf16* __restrict__ Bt,
                                                      float* __restrict__ C, int M) {
    const int lane = threadIdx.x & 31;
    const int wid  = blockIdx.x * (blockDim.x >> 5) + (threadIdx.x >> 5);
    const int ntiles64 = NDIM >> 6;
    const int tm = wid / ntiles64;
    const int tn = wid - tm * ntiles64;
    if (tm * 16 >= M) return;

    const int row  = tm * 16 + (lane & 15);
    const int ksel = (lane >> 4) << 4;   // A: lanes 16-31 hold K 16-31
    const int koff = (lane >> 4) << 4;   // B: lanes 16-31 hold K 16-31
    const bf16* __restrict__ arow = A + (size_t)row * KDIM;

    v8f acc[4] = {v8f{}, v8f{}, v8f{}, v8f{}};

    for (int k0 = 0; k0 < KDIM; k0 += 32) {
        // A fragment, contiguous K order matching B:
        //   lane<16 : halves 0-15 = K k0..k0+15
        //   lane>=16: halves 0-15 = K k0+16..k0+31
        v8bf alo = *(const v8bf*)(arow + k0 + ksel);
        v8bf ahi = *(const v8bf*)(arow + k0 + 8 + ksel);
        v16bf a;
#pragma unroll
        for (int i = 0; i < 8; ++i) { a[i] = alo[i]; a[8 + i] = ahi[i]; }

        const int colb = tn * 64 + (lane & 15);
#pragma unroll
        for (int nn = 0; nn < 4; ++nn) {
            // B fragment: column = lane&15, lanes 0-15 K k0..k0+15,
            // lanes 16-31 K k0+16..k0+31 -> contiguous 32B in Bt.
            v16bf b = *(const v16bf*)(Bt + (size_t)(colb + nn * 16) * KDIM + k0 + koff);
            acc[nn] = __builtin_amdgcn_wmma_f32_16x16x32_bf16(
                false, a, false, b, (short)0, acc[nn], false, false);
        }
    }

    // D layout: VGPR r -> lanes 0-15: M=r, lanes 16-31: M=r+8; N = lane&15.
    const int rbase = tm * 16 + ((lane >> 4) << 3);
#pragma unroll
    for (int nn = 0; nn < 4; ++nn) {
        const int col = tn * 64 + nn * 16 + (lane & 15);
#pragma unroll
        for (int r = 0; r < 8; ++r)
            C[(size_t)(rbase + r) * NDIM + col] = acc[nn][r];
    }
}

// ------------------------------------------------- aggregate (self-loop) init
// out[i,j] = bias[j] + perturb[i,j] + dinv[i]^2 * h[i,j]
template <int DIM>
__global__ void init_agg(const float* __restrict__ h, const float* __restrict__ dinv,
                         const float* __restrict__ bias, const float* __restrict__ perturb,
                         float* __restrict__ out, int n) {
    int g = blockIdx.x * blockDim.x + threadIdx.x;
    if (g >= n) return;
    int i = g / DIM;
    int j = g - i * DIM;
    float di = dinv[i];
    out[g] = bias[j] + perturb[g] + di * di * h[g];
}

// ------------------------------------------------------- edge scatter-add
// One thread = one (edge, 4-float chunk). h gather + atomic adds stay L2
// resident (h and out are ~20MB each vs 192MB L2).
template <int DIM>
__global__ void scatter_add(const int* __restrict__ src, const int* __restrict__ dst,
                            const float* __restrict__ dinv, const float* __restrict__ h,
                            float* __restrict__ out, int E) {
    const int CH = DIM / 4;
    int g = blockIdx.x * blockDim.x + threadIdx.x;
    if (g >= E * CH) return;
    int e = g / CH;
    int c = g - e * CH;
    int s = src[e];
    int d = dst[e];
    float w = dinv[s] * dinv[d];
    const float4 v = *(const float4*)(h + (size_t)s * DIM + (c << 2));
    float* o = out + (size_t)d * DIM + (c << 2);
    atomicAdd(o + 0, v.x * w);
    atomicAdd(o + 1, v.y * w);
    atomicAdd(o + 2, v.z * w);
    atomicAdd(o + 3, v.w * w);
}

// ---------------------------------------------------------------- launcher

extern "C" void kernel_launch(void* const* d_in, const int* in_sizes, int n_in,
                              void* d_out, int out_size, void* d_ws, size_t ws_size,
                              hipStream_t stream) {
    (void)n_in; (void)out_size; (void)ws_size;

    const float* x  = (const float*)d_in[0];
    const int*   ei = (const int*)d_in[1];
    const float* pf = (const float*)d_in[2];
    const float* pl = (const float*)d_in[3];
    const float* W1 = (const float*)d_in[4];
    const float* b1 = (const float*)d_in[5];
    const float* W2 = (const float*)d_in[6];
    const float* b2 = (const float*)d_in[7];
    const int E = in_sizes[1] / 2;
    const int* src = ei;
    const int* dst = ei + E;
    float* out = (float*)d_out;

    // workspace carve-out (256B aligned)
    char* ws = (char*)d_ws;
    size_t off = 0;
    auto alloc = [&](size_t bytes) -> char* {
        char* p = ws + off;
        off = (off + bytes + 255) & ~(size_t)255;
        return p;
    };
    float* dinv = (float*)alloc((size_t)N_NODES * 4);
    float* h1   = (float*)alloc((size_t)N_NODES * HID_DIM * 4);
    float* agg1 = (float*)alloc((size_t)N_NODES * HID_DIM * 4);
    float* h2   = (float*)alloc((size_t)N_NODES * OUT_DIM * 4);
    bf16*  xb   = (bf16*) alloc((size_t)N_NODES * IN_DIM * 2);
    bf16*  w1t  = (bf16*) alloc((size_t)IN_DIM * HID_DIM * 2);
    bf16*  a1b  = (bf16*) alloc((size_t)N_NODES * HID_DIM * 2);
    bf16*  w2t  = (bf16*) alloc((size_t)HID_DIM * OUT_DIM * 2);

    auto cdiv = [](int a, int b) { return (a + b - 1) / b; };

    // degree -> dinv
    fill_f32<<<cdiv(N_NODES, 256), 256, 0, stream>>>(dinv, 1.0f, N_NODES);
    deg_count<<<cdiv(E, 256), 256, 0, stream>>>(dst, dinv, E);
    rsqrt_inplace<<<cdiv(N_NODES, 256), 256, 0, stream>>>(dinv, N_NODES);

    // ---- layer 1: h1 = x @ W1 (bf16 WMMA, fp32 accum)
    cvt_bf16<<<cdiv(N_NODES * IN_DIM, 256), 256, 0, stream>>>(x, xb, N_NODES * IN_DIM);
    transpose_cvt<<<cdiv(IN_DIM * HID_DIM, 256), 256, 0, stream>>>(W1, w1t, IN_DIM, HID_DIM);
    {
        int waves = (N_NODES / 16) * (HID_DIM / 64);  // 5000
        gemm_bf16_wmma<IN_DIM, HID_DIM><<<cdiv(waves, 4), 128, 0, stream>>>(xb, w1t, h1, N_NODES);
    }
    init_agg<HID_DIM><<<cdiv(N_NODES * HID_DIM, 256), 256, 0, stream>>>(
        h1, dinv, b1, pf, agg1, N_NODES * HID_DIM);
    scatter_add<HID_DIM><<<cdiv(E * (HID_DIM / 4), 256), 256, 0, stream>>>(
        src, dst, dinv, h1, agg1, E);

    // ---- layer 2: h2 = agg1 @ W2
    cvt_bf16<<<cdiv(N_NODES * HID_DIM, 256), 256, 0, stream>>>(agg1, a1b, N_NODES * HID_DIM);
    transpose_cvt<<<cdiv(HID_DIM * OUT_DIM, 256), 256, 0, stream>>>(W2, w2t, HID_DIM, OUT_DIM);
    {
        int waves = (N_NODES / 16) * (OUT_DIM / 64);  // 2500
        gemm_bf16_wmma<HID_DIM, OUT_DIM><<<cdiv(waves, 4), 128, 0, stream>>>(a1b, w2t, h2, N_NODES);
    }
    init_agg<OUT_DIM><<<cdiv(N_NODES * OUT_DIM, 256), 256, 0, stream>>>(
        h2, dinv, b2, pl, out, N_NODES * OUT_DIM);
    scatter_add<OUT_DIM><<<cdiv(E * (OUT_DIM / 4), 256), 256, 0, stream>>>(
        src, dst, dinv, h2, out, E);
}
